// PaPi_11450382812118
// MI455X (gfx1250) — matrix operand order repaired and checked by reading, not compiled, not run on hardware
//
#include <hip/hip_runtime.h>
#include <hip/hip_bf16.h>
#include <math.h>

// ---------------------------------------------------------------------------
// MI455X (gfx1250) implementation of the PaPi reference.
// B=4096, DIN=2048, C=100, D=128.
// HBM-bound workload (~134 MB of f32 activations @ 23.3 TB/s ≈ 6 us floor):
//   - A tiles stream global->LDS via the Tensor Data Mover (tensor_load_to_lds,
//     TENSORcnt-synchronized, double-buffered) as raw f32,
//   - f32->f16 conversion happens at fragment build,
//   - matrix math on v_wmma_f32_16x16x32_f16 (f32 accumulate).
// ---------------------------------------------------------------------------

typedef __attribute__((ext_vector_type(16))) _Float16 v16h;
typedef __attribute__((ext_vector_type(8)))  _Float16 v8h;
typedef __attribute__((ext_vector_type(8)))  float    v8f;
typedef unsigned int u32x4 __attribute__((ext_vector_type(4)));
typedef int          i32x4 __attribute__((ext_vector_type(4)));
typedef int          i32x8 __attribute__((ext_vector_type(8)));

#define PROTO_W 0.99f

static constexpr int BM = 128, BN = 64, BK = 32;

// ---- Tensor Data Mover: load one BM x BK f32 tile (row-major, row stride K
// elements) from global memory into LDS at byte offset lds_off.
// D# layout per CDNA5 ISA (2-D tensor, groups 2/3 zero).
__device__ __forceinline__ void tdm_load_tile_f32(const float* gtile,
                                                  unsigned lds_off,
                                                  int K, int rowsLeft)
{
    const unsigned long long ga = (unsigned long long)(uintptr_t)gtile;
    u32x4 g0;
    g0.x = 0x1u;                                   // count=1 (valid), user mode
    g0.y = lds_off;                                // lds_addr [63:32]
    g0.z = (unsigned)(ga & 0xFFFFFFFFu);           // global_addr [95:64]
    g0.w = (unsigned)((ga >> 32) & 0x1FFFFFFu)     // global_addr [120:96]
         | (2u << 30);                             // type=2 ("image") [127:126]
    i32x8 g1;
    g1[0] = 0x20000;                               // data_size=2 (4 B), mask=0
    g1[1] = (int)(((unsigned)K & 0xFFFFu) << 16);  // tensor_dim0[15:0] @63:48
    g1[2] = (int)((((unsigned)K) >> 16)            // tensor_dim0[31:16]
         | (((unsigned)rowsLeft & 0xFFFFu) << 16));// tensor_dim1[15:0]
    g1[3] = (int)((((unsigned)rowsLeft) >> 16)     // tensor_dim1[31:16]
         | ((unsigned)BK << 16));                  // tile_dim0 = 32
    g1[4] = BM;                                    // tile_dim1 = 128, tile_dim2 = 0
    g1[5] = K;                                     // tensor_dim0_stride[31:0]
    g1[6] = 0;                                     // stride hi / dim1_stride lo
    g1[7] = 0;
    const i32x4 gz = {0, 0, 0, 0};
#if defined(__clang_major__) && (__clang_major__ >= 23)
    const i32x8 gz8 = {0, 0, 0, 0, 0, 0, 0, 0};
    __builtin_amdgcn_tensor_load_to_lds(g0, g1, gz, gz, gz8, 0);
#else
    __builtin_amdgcn_tensor_load_to_lds(g0, g1, gz, gz, 0);
#endif
}

// ---------------- WMMA GEMM ----------------
// Y[M x N] = A[M x K] @ B + (bias?), B is [K x N] (BT=0) or [N x K] (BT=1).
// 256 threads = 8 waves; tile 128(M) x 64(N); BK=32; wave grid 4(M) x 2(N);
// each wave owns 2x2 16x16 accumulators. Requires M%128==0, K%32==0.
__global__ __launch_bounds__(256)
void papi_gemm_f16_wmma(const float* __restrict__ A,
                        const float* __restrict__ Bm,
                        const float* __restrict__ bias,
                        float* __restrict__ Y,
                        int M, int K, int N, int BT)
{
    __shared__ float    As[2][BM][BK];   // 32 KB, filled by TDM (raw f32)
    __shared__ _Float16 Bst[2][BN][BK];  // 8 KB, (n, k) so frag reads are contiguous

    const int t     = threadIdx.x;
    const int lane  = t & 31;
    const int wave  = t >> 5;
    const int waveM = wave >> 1;         // 0..3 -> 32-row band
    const int waveN = wave & 1;          // 0..1 -> 32-col band
    const int rowBlk = blockIdx.y * BM;
    const int colBlk = blockIdx.x * BN;
    const int half = lane >> 4;
    const int l16  = lane & 15;
    // scalar (per-wave uniform) leader test -> real s_cbranch, since TDM
    // issues regardless of EXEC.
    const bool lead = (__builtin_amdgcn_readfirstlane(t) >> 5) == 0;

    const int T = K / BK;

    // Stage B tile kt into buffer b: 64 cols x 32 k, one ds_store_b128/thread.
    auto stageB = [&](int kt, int b) {
        const int ln = t >> 2;             // 0..63 column in tile
        const int lk = (t & 3) << 3;       // 0,8,16,24
        const int gn = colBlk + ln;
        const int gk = kt * BK + lk;
        v8h o;
        if (gn < N) {
            if (BT) {
                const float4 v0 = *(const float4*)(Bm + (size_t)gn * K + gk);
                const float4 v1 = *(const float4*)(Bm + (size_t)gn * K + gk + 4);
                o[0] = (_Float16)v0.x; o[1] = (_Float16)v0.y;
                o[2] = (_Float16)v0.z; o[3] = (_Float16)v0.w;
                o[4] = (_Float16)v1.x; o[5] = (_Float16)v1.y;
                o[6] = (_Float16)v1.z; o[7] = (_Float16)v1.w;
            } else {
                float tmp[8];
                #pragma unroll
                for (int i = 0; i < 8; ++i) tmp[i] = Bm[(size_t)(gk + i) * N + gn];
                #pragma unroll
                for (int i = 0; i < 8; ++i) o[i] = (_Float16)tmp[i];
            }
        } else {
            #pragma unroll
            for (int i = 0; i < 8; ++i) o[i] = (_Float16)0.0f;
        }
        *(v8h*)&Bst[b][ln][lk] = o;
    };

    // Prologue: kick off tile 0.
    if (lead)
        tdm_load_tile_f32(A + (size_t)rowBlk * K,
                          (unsigned)(uintptr_t)&As[0][0][0], K, M - rowBlk);
    stageB(0, 0);

    v8f acc[2][2] = {};

    for (int kt = 0; kt < T; ++kt) {
        const int cur = kt & 1, nxt = cur ^ 1;
        if (kt + 1 < T) {
            if (lead)
                tdm_load_tile_f32(A + (size_t)rowBlk * K + (size_t)(kt + 1) * BK,
                                  (unsigned)(uintptr_t)&As[nxt][0][0], K, M - rowBlk);
            stageB(kt + 1, nxt);
            __builtin_amdgcn_s_wait_tensorcnt((short)1);  // tile kt landed
        } else {
            __builtin_amdgcn_s_wait_tensorcnt((short)0);
        }
        __syncthreads();   // A(kt) + B(kt) visible to all waves

        // ---- fragments (ISA 7.12.2 layouts); f32->f16 at build time ----
        v16h afrag[2], bfrag[2];
        #pragma unroll
        for (int i = 0; i < 2; ++i) {
            // A 16x32: elems 0..7 -> K=half*8+e; elems 8..15 -> K=16+half*8+e.
            const float* rp = &As[cur][waveM * 32 + i * 16 + l16][0];
            const float4 a0 = *(const float4*)(rp + half * 8);
            const float4 a1 = *(const float4*)(rp + half * 8 + 4);
            const float4 a2 = *(const float4*)(rp + 16 + half * 8);
            const float4 a3 = *(const float4*)(rp + 16 + half * 8 + 4);
            v16h f;
            f[0]  = (_Float16)a0.x; f[1]  = (_Float16)a0.y;
            f[2]  = (_Float16)a0.z; f[3]  = (_Float16)a0.w;
            f[4]  = (_Float16)a1.x; f[5]  = (_Float16)a1.y;
            f[6]  = (_Float16)a1.z; f[7]  = (_Float16)a1.w;
            f[8]  = (_Float16)a2.x; f[9]  = (_Float16)a2.y;
            f[10] = (_Float16)a2.z; f[11] = (_Float16)a2.w;
            f[12] = (_Float16)a3.x; f[13] = (_Float16)a3.y;
            f[14] = (_Float16)a3.z; f[15] = (_Float16)a3.w;
            afrag[i] = f;
        }
        #pragma unroll
        for (int j = 0; j < 2; ++j) {
            // B 32x16: elems 0..7 -> K=half*8+e; elems 8..15 -> K=16+half*8+e.
            const int c = waveN * 32 + j * 16 + l16;
            v8h lo = *(const v8h*)&Bst[cur][c][half * 8];
            v8h hi = *(const v8h*)&Bst[cur][c][16 + half * 8];
            bfrag[j] = __builtin_shufflevector(lo, hi,
                        0,1,2,3,4,5,6,7,8,9,10,11,12,13,14,15);
        }
        #pragma unroll
        for (int i = 0; i < 2; ++i)
            #pragma unroll
            for (int j = 0; j < 2; ++j)
                acc[i][j] = __builtin_amdgcn_wmma_f32_16x16x32_f16(
                    false, afrag[i], false, bfrag[j],
                    (short)0, acc[i][j], false, false);

        __syncthreads();   // everyone done with buffer `cur` before reuse
    }

    // ---- epilogue: C/D layout: VGPR v -> row = v + half*8; col = l16 ----
    #pragma unroll
    for (int i = 0; i < 2; ++i) {
        #pragma unroll
        for (int j = 0; j < 2; ++j) {
            const int col = colBlk + waveN * 32 + j * 16 + l16;
            if (col >= N) continue;
            const float bv = bias ? bias[col] : 0.0f;
            const int rowBase = rowBlk + waveM * 32 + i * 16 + half * 8;
            #pragma unroll
            for (int v = 0; v < 8; ++v)
                Y[(size_t)(rowBase + v) * N + col] = acc[i][j][v] + bv;
        }
    }
}

// ---------------- row-wise L2 normalize (cols == 128), in place ----------------
__global__ __launch_bounds__(256)
void papi_l2norm_rows(float* __restrict__ X, int rows)
{
    const int wave = threadIdx.x >> 5, lane = threadIdx.x & 31;
    const int row = blockIdx.x * 8 + wave;
    if (row >= rows) return;
    float4* p = (float4*)(X + (size_t)row * 128);
    float4 v = p[lane];
    float ss = v.x * v.x + v.y * v.y + v.z * v.z + v.w * v.w;
    #pragma unroll
    for (int m = 16; m; m >>= 1) ss += __shfl_xor(ss, m, 32);
    const float s = 1.0f / fmaxf(sqrtf(ss), 1e-12f);
    v.x *= s; v.y *= s; v.z *= s; v.w *= s;
    p[lane] = v;
}

// ---------------- masked-softmax argmax (pseudo labels) ----------------
// argmax_c softmax(logits)[c]*Y[c] == argmax_c exp(logits[c]-max)*Y[c]
__global__ __launch_bounds__(256)
void papi_pseudo_label(const float* __restrict__ logits,
                       const float* __restrict__ Ymask,
                       int* __restrict__ out, int rows, int C)
{
    const int wave = threadIdx.x >> 5, lane = threadIdx.x & 31;
    const int row = blockIdx.x * 8 + wave;
    if (row >= rows) return;
    const float* lp = logits + (size_t)row * C;
    const float* yp = Ymask  + (size_t)row * C;
    float m = -INFINITY;
    for (int c = lane; c < C; c += 32) m = fmaxf(m, lp[c]);
    #pragma unroll
    for (int s = 16; s; s >>= 1) m = fmaxf(m, __shfl_xor(m, s, 32));
    float best = -1.0f; int bidx = 0x7fffffff;
    for (int c = lane; c < C; c += 32) {
        const float v = __expf(lp[c] - m) * yp[c];
        if (v > best) { best = v; bidx = c; }
    }
    #pragma unroll
    for (int s = 16; s; s >>= 1) {
        const float ob = __shfl_xor(best, s, 32);
        const int   oi = __shfl_xor(bidx, s, 32);
        if (ob > best || (ob == best && oi < bidx)) { best = ob; bidx = oi; }
    }
    if (lane == 0) out[row] = bidx;
}

// ---------------- per-class sequential EMA + L2 norm ----------------
// Classes are independent in the reference scan; block c replays the 2B-sample
// sequence (q then k) for label c only. D == 128 == blockDim.x.
__global__ __launch_bounds__(128)
void papi_proto_ema(const float* __restrict__ protos,
                    const float* __restrict__ fq, const float* __restrict__ fk,
                    const int* __restrict__ lq, const int* __restrict__ lk,
                    float* __restrict__ outp, int Bn)
{
    const int c = blockIdx.x;
    const int d = threadIdx.x;
    float acc = protos[(size_t)c * 128 + d];
    for (int i = 0; i < Bn; ++i)
        if (lq[i] == c) acc = PROTO_W * acc + (1.0f - PROTO_W) * fq[(size_t)i * 128 + d];
    for (int i = 0; i < Bn; ++i)
        if (lk[i] == c) acc = PROTO_W * acc + (1.0f - PROTO_W) * fk[(size_t)i * 128 + d];

    __shared__ float red[128];
    red[d] = acc * acc;
    __syncthreads();
    #pragma unroll
    for (int s = 64; s >= 1; s >>= 1) {
        if (d < s) red[d] += red[d + s];
        __syncthreads();
    }
    outp[(size_t)c * 128 + d] = acc / fmaxf(sqrtf(red[0]), 1e-12f);
}

// ---------------- host orchestration ----------------
extern "C" void kernel_launch(void* const* d_in, const int* in_sizes, int n_in,
                              void* d_out, int out_size, void* d_ws, size_t ws_size,
                              hipStream_t stream)
{
    const float* img_q  = (const float*)d_in[0];
    const float* img_k  = (const float*)d_in[1];
    const float* img_qm = (const float*)d_in[2];
    const float* img_km = (const float*)d_in[3];
    const float* partY  = (const float*)d_in[4];
    const float* W_cls  = (const float*)d_in[5];
    const float* b_cls  = (const float*)d_in[6];
    const float* W_proj = (const float*)d_in[7];
    const float* protos = (const float*)d_in[8];

    const int B = 4096, DIN = 2048, C = 100, D = 128;

    float* out  = (float*)d_out;
    float* out_q = out;
    float* out_k = out + (size_t)B * C;
    float* lp_q  = out + 2 * (size_t)B * C;
    float* lp_k  = out + 3 * (size_t)B * C;
    float* lp_qm = out + 4 * (size_t)B * C;
    float* lp_km = out + 5 * (size_t)B * C;
    float* newp  = out + 6 * (size_t)B * C;

    char* ws = (char*)d_ws;
    float* pq  = (float*)ws; ws += (size_t)B * D * sizeof(float);
    float* pk  = (float*)ws; ws += (size_t)B * D * sizeof(float);
    float* pqm = (float*)ws; ws += (size_t)B * D * sizeof(float);
    float* pkm = (float*)ws; ws += (size_t)B * D * sizeof(float);
    int* lblq = (int*)ws; ws += (size_t)B * sizeof(int);
    int* lblk = (int*)ws; ws += (size_t)B * sizeof(int);

    const dim3 blk(256);
    const dim3 gCls((C + BN - 1) / BN, B / BM);   // (2, 32)
    const dim3 gPrj((D + BN - 1) / BN, B / BM);   // (2, 32)

    // classifier heads (only q, k are consumed by the reference outputs)
    papi_gemm_f16_wmma<<<gCls, blk, 0, stream>>>(img_q, W_cls, b_cls, out_q, B, DIN, C, 0);
    papi_gemm_f16_wmma<<<gCls, blk, 0, stream>>>(img_k, W_cls, b_cls, out_k, B, DIN, C, 0);

    // projection heads (all four)
    papi_gemm_f16_wmma<<<gPrj, blk, 0, stream>>>(img_q,  W_proj, nullptr, pq,  B, DIN, D, 0);
    papi_gemm_f16_wmma<<<gPrj, blk, 0, stream>>>(img_k,  W_proj, nullptr, pk,  B, DIN, D, 0);
    papi_gemm_f16_wmma<<<gPrj, blk, 0, stream>>>(img_qm, W_proj, nullptr, pqm, B, DIN, D, 0);
    papi_gemm_f16_wmma<<<gPrj, blk, 0, stream>>>(img_km, W_proj, nullptr, pkm, B, DIN, D, 0);

    papi_l2norm_rows<<<B / 8, 256, 0, stream>>>(pq,  B);
    papi_l2norm_rows<<<B / 8, 256, 0, stream>>>(pk,  B);
    papi_l2norm_rows<<<B / 8, 256, 0, stream>>>(pqm, B);
    papi_l2norm_rows<<<B / 8, 256, 0, stream>>>(pkm, B);

    papi_pseudo_label<<<B / 8, 256, 0, stream>>>(out_q, partY, lblq, B, C);
    papi_pseudo_label<<<B / 8, 256, 0, stream>>>(out_k, partY, lblk, B, C);

    // prototype logits: proj_norm @ protos^T (protos is [C x D] row-major -> BT=1)
    papi_gemm_f16_wmma<<<gCls, blk, 0, stream>>>(pq,  protos, nullptr, lp_q,  B, D, C, 1);
    papi_gemm_f16_wmma<<<gCls, blk, 0, stream>>>(pk,  protos, nullptr, lp_k,  B, D, C, 1);
    papi_gemm_f16_wmma<<<gCls, blk, 0, stream>>>(pqm, protos, nullptr, lp_qm, B, D, C, 1);
    papi_gemm_f16_wmma<<<gCls, blk, 0, stream>>>(pkm, protos, nullptr, lp_km, B, D, C, 1);

    // sequential EMA scatter (per-class exact replay) + final L2 norm
    papi_proto_ema<<<C, 128, 0, stream>>>(protos, pq, pk, lblq, lblk, newp, B);
}
